// SwinTransformer_82858509075002
// MI455X (gfx1250) — matrix operand order repaired and checked by reading, not compiled, not run on hardware
//
#include <hip/hip_runtime.h>
#include <hip/hip_bf16.h>
#include <math.h>

typedef __attribute__((ext_vector_type(16))) _Float16 v16h;
typedef __attribute__((ext_vector_type(8)))  float    v8f;

#define LN_EPS 1e-5f
#define CDIM   96
#define C3     288
#define NTOK   64     // tokens per 4x4x4 window
#define NHEAD  3
#define HDIM   32

// Swizzled f16 weight arena layout (per block, units = halves):
//   qkv_w  : off 0      KT=3  NT=18  (27648)
//   proj_w : off 27648  KT=3  NT=6   ( 9216)
//   mlp_w1 : off 36864  KT=3  NT=24  (36864)
//   mlp_w2 : off 73728  KT=12 NT=6   (36864)
// per-block total 110592 halves; tile = 512 halves, order (nt*KT + kt),
// within tile: lane*16 + h, element = w[(kt*32 + (lane>>4)*16 + h)*ldn + nt*16 + (lane&15)]
#define SWZ_QKV   0
#define SWZ_PROJ  27648
#define SWZ_W1    36864
#define SWZ_W2    73728
#define SWZ_BLOCK 110592
#define SWZ_BYTES (2u * SWZ_BLOCK * 2u)

// ---------------- WMMA helpers (gfx1250, wave32) ----------------

__device__ __forceinline__ v8f wmma_f16(v16h a, v16h b, v8f c) {
  // D = A(16x32 f16) x B(32x16 f16) + C(16x16 f32)
  return __builtin_amdgcn_wmma_f32_16x16x32_f16(
      false, a, false, b, (short)0, c, false, false);
}

// A fragment from row-major f16 LDS: A(m,k) = s[(m0+m)*ld + k0+k]
// A 16x32: m = lane&15, kb=(lane>>4)*8, half h -> K=2*kb+h (contiguous, matches B)
__device__ __forceinline__ v16h frag_a_f16(const _Float16* s, int m0, int ld,
                                           int k0, int rn, int kb) {
  v16h a;
#pragma unroll
  for (int h = 0; h < 16; ++h)
    a[h] = s[(m0 + rn) * ld + k0 + 2 * kb + h];
  return a;
}

// A fragment from row-major f32 LDS (softmax probabilities), convert to f16.
__device__ __forceinline__ v16h frag_a_f32(const float* s, int m0, int ld,
                                           int k0, int rn, int kb) {
  v16h a;
#pragma unroll
  for (int h = 0; h < 16; ++h)
    a[h] = (_Float16)s[(m0 + rn) * ld + k0 + 2 * kb + h];
  return a;
}

// B fragment from pre-swizzled f16 arena: one aligned 32B load per lane.
__device__ __forceinline__ v16h frag_b_swz(const _Float16* wz, int tile, int lane) {
  return *(const v16h*)(wz + (size_t)tile * 512 + lane * 16);
}

// Fallback: B fragment from row-major [K][N] f32 global weights.
__device__ __forceinline__ v16h frag_b_glb(const float* w, int k0, int ldn,
                                           int n0, int rn, int kh) {
  v16h b;
#pragma unroll
  for (int h = 0; h < 16; ++h)
    b[h] = (_Float16)w[(k0 + kh + h) * ldn + n0 + rn];
  return b;
}

// B fragment from row-major [K][N] f16 LDS (V matrix).
__device__ __forceinline__ v16h frag_b_lds(const _Float16* s, int k0, int ld,
                                           int n0, int rn, int kh) {
  v16h b;
#pragma unroll
  for (int h = 0; h < 16; ++h)
    b[h] = s[(k0 + kh + h) * ld + n0 + rn];
  return b;
}

// B^T fragment: B(k,n) = s[(n0+n)*ld + k0+k]  (K matrix for Q.K^T)
__device__ __forceinline__ v16h frag_bt_lds(const _Float16* s, int n0, int ld,
                                            int k0, int rn, int kh) {
  v16h b;
#pragma unroll
  for (int h = 0; h < 16; ++h)
    b[h] = s[(n0 + rn) * ld + k0 + kh + h];
  return b;
}

// token t in window -> flat element base in (2,64,64,64,96), undoing the shift roll
__device__ __forceinline__ long tok_base(int b, int wd, int wh, int ww,
                                         int t, int shift) {
  int td = t >> 4, th = (t >> 2) & 3, tw = t & 3;
  int d = (wd * 4 + td + shift) & 63;
  int h = (wh * 4 + th + shift) & 63;
  int w = (ww * 4 + tw + shift) & 63;
  return ((((long)b * 64 + d) * 64 + h) * 64 + w) * (long)CDIM;
}

__device__ __forceinline__ int rel_index(int q, int k) {
  int dq = q >> 4, hq = (q >> 2) & 3, wq = q & 3;
  int dk = k >> 4, hk = (k >> 2) & 3, wk = k & 3;
  return (dq - dk + 3) * 49 + (hq - hk + 3) * 7 + (wq - wk + 3);
}

// ---------------- Weight prep: f32 row-major -> f16 B-fragment-swizzled ----------

__global__ __launch_bounds__(256)
void swin_prep_weights(const float* __restrict__ qkv_w, const float* __restrict__ proj_w,
                       const float* __restrict__ w1, const float* __restrict__ w2,
                       _Float16* __restrict__ out) {
  int idx = blockIdx.x * blockDim.x + threadIdx.x;
  if (idx >= 2 * SWZ_BLOCK) return;
  int blk = idx / SWZ_BLOCK;
  int rem = idx - blk * SWZ_BLOCK;
  const float* src; int KT, ldn, off;
  if (rem < SWZ_PROJ)      { src = qkv_w  + (size_t)blk * CDIM * C3;       off = rem;            KT = 3;  ldn = C3;       }
  else if (rem < SWZ_W1)   { src = proj_w + (size_t)blk * CDIM * CDIM;     off = rem - SWZ_PROJ; KT = 3;  ldn = CDIM;     }
  else if (rem < SWZ_W2)   { src = w1     + (size_t)blk * CDIM * 4 * CDIM; off = rem - SWZ_W1;   KT = 3;  ldn = 4 * CDIM; }
  else                     { src = w2     + (size_t)blk * 4 * CDIM * CDIM; off = rem - SWZ_W2;   KT = 12; ldn = CDIM;     }
  int tile   = off >> 9;
  int within = off & 511;
  int lane = within >> 4;
  int h    = within & 15;
  int nt = tile / KT;
  int kt = tile - nt * KT;
  int k = kt * 32 + (lane >> 4) * 16 + h;
  int n = nt * 16 + (lane & 15);
  out[idx] = (_Float16)src[(size_t)k * ldn + n];
}

// ---------------- Fused windowed attention (LN1 + QKV + attn + proj + residual) ----

template <bool SWZ>
__global__ __launch_bounds__(128)
void swin_attn_kernel(const float* __restrict__ xin, float* __restrict__ yout,
                      const float* __restrict__ qkv_w, const float* __restrict__ qkv_b,
                      const float* __restrict__ proj_w, const float* __restrict__ proj_b,
                      const _Float16* __restrict__ wz,     // swizzled arena (this block)
                      const float* __restrict__ table,
                      const float* __restrict__ ln_g, const float* __restrict__ ln_b,
                      int shift) {
  // 64 KB LDS, phase-aliased:
  //  [0,12288)      : xn (LN output, f16 64x96)  -> reused as attn-out (f16 64x96)
  //  [12288,49152)  : qkv f16 64x288 (q:0..95 scaled, k:96..191, v:192..287)
  //  [49152,65536)  : per-head scores f32 64x64
  __shared__ __align__(16) char smem[65536];
  _Float16* xn  = (_Float16*)smem;
  _Float16* qkv = (_Float16*)(smem + 12288);
  float*    sc  = (float*)(smem + 49152);
  _Float16* oh  = (_Float16*)smem;   // aliases xn after phase 2

  const int tid  = threadIdx.x;
  const int lane = tid & 31;
  const int wv   = tid >> 5;
  const int rn   = lane & 15;
  const int kb   = (lane >> 4) * 8;    // A-frag K base & C-frag M base
  const int kh   = (lane >> 4) * 16;   // B-frag K base

  const int wid  = blockIdx.x;
  const int b    = wid >> 12;
  const int wrem = wid & 4095;
  const int wd = wrem >> 8, wh = (wrem >> 4) & 15, ww = wrem & 15;

  // ---- Phase 1: LayerNorm per token into f16 LDS (float4 vector loads)
  if (tid < NTOK) {
    long base = tok_base(b, wd, wh, ww, tid, shift);
    const float4* r4 = (const float4*)(xin + base);
    float s0 = 0.f, s1 = 0.f;
#pragma unroll 4
    for (int c4 = 0; c4 < CDIM / 4; ++c4) {
      float4 v = r4[c4];
      s0 += v.x + v.y + v.z + v.w;
      s1 += v.x * v.x + v.y * v.y + v.z * v.z + v.w * v.w;
    }
    float mean = s0 * (1.0f / CDIM);
    float var  = s1 * (1.0f / CDIM) - mean * mean;
    float rs   = rsqrtf(var + LN_EPS);
#pragma unroll 4
    for (int c4 = 0; c4 < CDIM / 4; ++c4) {
      float4 v = r4[c4];
      int c = c4 * 4;
      xn[tid * CDIM + c + 0] = (_Float16)((v.x - mean) * rs * ln_g[c + 0] + ln_b[c + 0]);
      xn[tid * CDIM + c + 1] = (_Float16)((v.y - mean) * rs * ln_g[c + 1] + ln_b[c + 1]);
      xn[tid * CDIM + c + 2] = (_Float16)((v.z - mean) * rs * ln_g[c + 2] + ln_b[c + 2]);
      xn[tid * CDIM + c + 3] = (_Float16)((v.w - mean) * rs * ln_g[c + 3] + ln_b[c + 3]);
    }
  }
  __syncthreads();

  // ---- Phase 2: QKV = xn(64x96) @ qkv_w(96x288) + bias ; q scaled by 1/sqrt(hd)
  for (int nt = wv; nt < 18; nt += 4) {
    const int n0 = nt * 16;
    v8f acc[4];
    const v8f zero = {0.f, 0.f, 0.f, 0.f, 0.f, 0.f, 0.f, 0.f};
#pragma unroll
    for (int mt = 0; mt < 4; ++mt) acc[mt] = zero;
#pragma unroll
    for (int kt = 0; kt < 3; ++kt) {
      v16h bf = SWZ ? frag_b_swz(wz + SWZ_QKV, nt * 3 + kt, lane)
                    : frag_b_glb(qkv_w, kt * 32, C3, n0, rn, kh);
#pragma unroll
      for (int mt = 0; mt < 4; ++mt) {
        v16h af = frag_a_f16(xn, mt * 16, CDIM, kt * 32, rn, kb);
        acc[mt] = wmma_f16(af, bf, acc[mt]);
      }
    }
    const int   col  = n0 + rn;
    const float bias = qkv_b[col];
    const float scl  = (col < CDIM) ? 0.17677669529663689f : 1.0f; // 32^-0.5
#pragma unroll
    for (int mt = 0; mt < 4; ++mt)
#pragma unroll
      for (int r = 0; r < 8; ++r) {
        int m = mt * 16 + kb + r;
        qkv[m * C3 + col] = (_Float16)((acc[mt][r] + bias) * scl);
      }
  }
  __syncthreads();

  // ---- Phase 3: per-head attention
  for (int head = 0; head < NHEAD; ++head) {
    const int qo = head * HDIM;            // q columns
    const int ko = CDIM + head * HDIM;     // k columns
    const int vo = 2 * CDIM + head * HDIM; // v columns

    // scores S = q @ k^T  (4x4 tiles, one N-tile per wave, K=32 -> 1 WMMA)
    {
      const int nt = wv;
      v16h bf = frag_bt_lds(qkv, nt * 16, C3, ko, rn, kh);
#pragma unroll
      for (int mt = 0; mt < 4; ++mt) {
        v16h af = frag_a_f16(qkv, mt * 16, C3, qo, rn, kb);
        v8f c = {0.f, 0.f, 0.f, 0.f, 0.f, 0.f, 0.f, 0.f};
        c = wmma_f16(af, bf, c);
#pragma unroll
        for (int r = 0; r < 8; ++r) {
          int qt = mt * 16 + kb + r;
          int kt2 = nt * 16 + rn;
          float bias = table[rel_index(qt, kt2) * NHEAD + head];
          sc[qt * NTOK + kt2] = c[r] + bias;
        }
      }
    }
    __syncthreads();

    // softmax over keys (one row per thread, fp32)
    if (tid < NTOK) {
      float* row = sc + tid * NTOK;
      float mx = -3.4e38f;
#pragma unroll 4
      for (int k = 0; k < NTOK; ++k) mx = fmaxf(mx, row[k]);
      float sum = 0.f;
#pragma unroll 4
      for (int k = 0; k < NTOK; ++k) { float e = __expf(row[k] - mx); row[k] = e; sum += e; }
      float inv = 1.0f / sum;
#pragma unroll 4
      for (int k = 0; k < NTOK; ++k) row[k] *= inv;
    }
    __syncthreads();

    // out_h = P(64x64) @ V_h(64x32)   (4x2 tiles -> 2 per wave, K=64 -> 2 WMMAs)
#pragma unroll
    for (int ti = wv * 2; ti < wv * 2 + 2; ++ti) {
      const int mt = ti >> 1, nt = ti & 1;
      v8f c = {0.f, 0.f, 0.f, 0.f, 0.f, 0.f, 0.f, 0.f};
#pragma unroll
      for (int kt = 0; kt < 2; ++kt) {
        v16h af = frag_a_f32(sc, mt * 16, NTOK, kt * 32, rn, kb);
        v16h bf = frag_b_lds(qkv, kt * 32, C3, vo + nt * 16, rn, kh);
        c = wmma_f16(af, bf, c);
      }
#pragma unroll
      for (int r = 0; r < 8; ++r) {
        int m = mt * 16 + kb + r;
        oh[m * CDIM + qo + nt * 16 + rn] = (_Float16)c[r];
      }
    }
    __syncthreads();   // protect sc before next head rewrites it
  }

  // ---- Phase 4: projection + residual + (shifted) window-reverse store
  for (int nt = wv; nt < 6; nt += 4) {
    const int n0 = nt * 16;
    v8f acc[4];
    const v8f zero = {0.f, 0.f, 0.f, 0.f, 0.f, 0.f, 0.f, 0.f};
#pragma unroll
    for (int mt = 0; mt < 4; ++mt) acc[mt] = zero;
#pragma unroll
    for (int kt = 0; kt < 3; ++kt) {
      v16h bf = SWZ ? frag_b_swz(wz + SWZ_PROJ, nt * 3 + kt, lane)
                    : frag_b_glb(proj_w, kt * 32, CDIM, n0, rn, kh);
#pragma unroll
      for (int mt = 0; mt < 4; ++mt) {
        v16h af = frag_a_f16(oh, mt * 16, CDIM, kt * 32, rn, kb);
        acc[mt] = wmma_f16(af, bf, acc[mt]);
      }
    }
    const int   col = n0 + rn;
    const float pb  = proj_b[col];
#pragma unroll
    for (int mt = 0; mt < 4; ++mt)
#pragma unroll
      for (int r = 0; r < 8; ++r) {
        int tok = mt * 16 + kb + r;
        long base = tok_base(b, wd, wh, ww, tok, shift);
        yout[base + col] = xin[base + col] + acc[mt][r] + pb;
      }
  }
}

// ---------------- Fused MLP (LN2 + GELU MLP + residual, in-place) ----------------

template <bool SWZ>
__global__ __launch_bounds__(128)
void swin_mlp_kernel(float* __restrict__ y,
                     const float* __restrict__ ln_g, const float* __restrict__ ln_b,
                     const float* __restrict__ w1, const float* __restrict__ b1,
                     const float* __restrict__ w2, const float* __restrict__ b2,
                     const _Float16* __restrict__ wz) {
  __shared__ __align__(16) char smem[61440];
  _Float16* xn = (_Float16*)smem;            // 64 x 96 f16
  _Float16* h1 = (_Float16*)(smem + 12288);  // 64 x 384 f16

  const int tid  = threadIdx.x;
  const int lane = tid & 31;
  const int wv   = tid >> 5;
  const int rn   = lane & 15;
  const int kb   = (lane >> 4) * 8;
  const int kh   = (lane >> 4) * 16;

  const long base = (long)blockIdx.x * NTOK * CDIM;

  // LN2 per token (float4 vector loads)
  if (tid < NTOK) {
    const float4* r4 = (const float4*)(y + base + (long)tid * CDIM);
    float s0 = 0.f, s1 = 0.f;
#pragma unroll 4
    for (int c4 = 0; c4 < CDIM / 4; ++c4) {
      float4 v = r4[c4];
      s0 += v.x + v.y + v.z + v.w;
      s1 += v.x * v.x + v.y * v.y + v.z * v.z + v.w * v.w;
    }
    float mean = s0 * (1.0f / CDIM);
    float var  = s1 * (1.0f / CDIM) - mean * mean;
    float rs   = rsqrtf(var + LN_EPS);
#pragma unroll 4
    for (int c4 = 0; c4 < CDIM / 4; ++c4) {
      float4 v = r4[c4];
      int c = c4 * 4;
      xn[tid * CDIM + c + 0] = (_Float16)((v.x - mean) * rs * ln_g[c + 0] + ln_b[c + 0]);
      xn[tid * CDIM + c + 1] = (_Float16)((v.y - mean) * rs * ln_g[c + 1] + ln_b[c + 1]);
      xn[tid * CDIM + c + 2] = (_Float16)((v.z - mean) * rs * ln_g[c + 2] + ln_b[c + 2]);
      xn[tid * CDIM + c + 3] = (_Float16)((v.w - mean) * rs * ln_g[c + 3] + ln_b[c + 3]);
    }
  }
  __syncthreads();

  // GEMM1: (64x96) @ w1(96x384) + b1, exact GELU, -> h1 f16
  for (int nt = wv; nt < 24; nt += 4) {
    const int n0 = nt * 16;
    v8f acc[4];
    const v8f zero = {0.f, 0.f, 0.f, 0.f, 0.f, 0.f, 0.f, 0.f};
#pragma unroll
    for (int mt = 0; mt < 4; ++mt) acc[mt] = zero;
#pragma unroll
    for (int kt = 0; kt < 3; ++kt) {
      v16h bf = SWZ ? frag_b_swz(wz + SWZ_W1, nt * 3 + kt, lane)
                    : frag_b_glb(w1, kt * 32, 4 * CDIM, n0, rn, kh);
#pragma unroll
      for (int mt = 0; mt < 4; ++mt) {
        v16h af = frag_a_f16(xn, mt * 16, CDIM, kt * 32, rn, kb);
        acc[mt] = wmma_f16(af, bf, acc[mt]);
      }
    }
    const int   col = n0 + rn;
    const float bb  = b1[col];
#pragma unroll
    for (int mt = 0; mt < 4; ++mt)
#pragma unroll
      for (int r = 0; r < 8; ++r) {
        int m = mt * 16 + kb + r;
        float v = acc[mt][r] + bb;
        float g = 0.5f * v * (1.0f + erff(v * 0.70710678118654752f)); // exact GELU
        h1[m * (4 * CDIM) + col] = (_Float16)g;
      }
  }
  __syncthreads();

  // GEMM2: (64x384) @ w2(384x96) + b2 + residual, in-place store
  for (int nt = wv; nt < 6; nt += 4) {
    const int n0 = nt * 16;
    v8f acc[4];
    const v8f zero = {0.f, 0.f, 0.f, 0.f, 0.f, 0.f, 0.f, 0.f};
#pragma unroll
    for (int mt = 0; mt < 4; ++mt) acc[mt] = zero;
#pragma unroll
    for (int kt = 0; kt < 12; ++kt) {
      v16h bf = SWZ ? frag_b_swz(wz + SWZ_W2, nt * 12 + kt, lane)
                    : frag_b_glb(w2, kt * 32, CDIM, n0, rn, kh);
#pragma unroll
      for (int mt = 0; mt < 4; ++mt) {
        v16h af = frag_a_f16(h1, mt * 16, 4 * CDIM, kt * 32, rn, kb);
        acc[mt] = wmma_f16(af, bf, acc[mt]);
      }
    }
    const int   col = n0 + rn;
    const float bb  = b2[col];
#pragma unroll
    for (int mt = 0; mt < 4; ++mt)
#pragma unroll
      for (int r = 0; r < 8; ++r) {
        int tok = mt * 16 + kb + r;
        long idx = base + (long)tok * CDIM + col;
        y[idx] = y[idx] + acc[mt][r] + bb;
      }
  }
}

// ---------------- launch ----------------

extern "C" void kernel_launch(void* const* d_in, const int* in_sizes, int n_in,
                              void* d_out, int out_size, void* d_ws, size_t ws_size,
                              hipStream_t stream) {
  const float* x      = (const float*)d_in[0];
  const float* qkv_w  = (const float*)d_in[1];
  const float* qkv_b  = (const float*)d_in[2];
  const float* proj_w = (const float*)d_in[3];
  const float* proj_b = (const float*)d_in[4];
  const float* table  = (const float*)d_in[5];
  const float* ln1g   = (const float*)d_in[6];
  const float* ln1b   = (const float*)d_in[7];
  const float* ln2g   = (const float*)d_in[8];
  const float* ln2b   = (const float*)d_in[9];
  const float* w1     = (const float*)d_in[10];
  const float* b1     = (const float*)d_in[11];
  const float* w2     = (const float*)d_in[12];
  const float* b2     = (const float*)d_in[13];
  float* out = (float*)d_out;

  const int NWIN  = 2 * 16 * 16 * 16;            // 8192 windows
  const int NTILE = (2 * 64 * 64 * 64) / NTOK;   // 8192 token tiles

  const bool use_swz = (ws_size >= (size_t)SWZ_BYTES);
  _Float16* wz = (_Float16*)d_ws;

  if (use_swz) {
    const int total = 2 * SWZ_BLOCK;
    swin_prep_weights<<<(total + 255) / 256, 256, 0, stream>>>(qkv_w, proj_w, w1, w2, wz);
  }

  for (int i = 0; i < 2; ++i) {
    const int shift = (i == 0) ? 0 : 2;
    const float* xi = (i == 0) ? x : out;  // attn1 runs in-place on d_out
    const _Float16* wzb = use_swz ? (wz + (size_t)i * SWZ_BLOCK) : (const _Float16*)nullptr;
    if (use_swz) {
      swin_attn_kernel<true><<<NWIN, 128, 0, stream>>>(
          xi, out,
          qkv_w + (size_t)i * CDIM * C3, qkv_b + (size_t)i * C3,
          proj_w + (size_t)i * CDIM * CDIM, proj_b + (size_t)i * CDIM,
          wzb, table + (size_t)i * 343 * NHEAD,
          ln1g + (size_t)i * CDIM, ln1b + (size_t)i * CDIM, shift);
      swin_mlp_kernel<true><<<NTILE, 128, 0, stream>>>(
          out, ln2g + (size_t)i * CDIM, ln2b + (size_t)i * CDIM,
          w1 + (size_t)i * CDIM * 4 * CDIM, b1 + (size_t)i * 4 * CDIM,
          w2 + (size_t)i * 4 * CDIM * CDIM, b2 + (size_t)i * CDIM, wzb);
    } else {
      swin_attn_kernel<false><<<NWIN, 128, 0, stream>>>(
          xi, out,
          qkv_w + (size_t)i * CDIM * C3, qkv_b + (size_t)i * C3,
          proj_w + (size_t)i * CDIM * CDIM, proj_b + (size_t)i * CDIM,
          wzb, table + (size_t)i * 343 * NHEAD,
          ln1g + (size_t)i * CDIM, ln1b + (size_t)i * CDIM, shift);
      swin_mlp_kernel<false><<<NTILE, 128, 0, stream>>>(
          out, ln2g + (size_t)i * CDIM, ln2b + (size_t)i * CDIM,
          w1 + (size_t)i * CDIM * 4 * CDIM, b1 + (size_t)i * 4 * CDIM,
          w2 + (size_t)i * 4 * CDIM * CDIM, b2 + (size_t)i * CDIM, wzb);
    }
  }
}
